// MDAUnit_23201413333219
// MI455X (gfx1250) — matrix operand order repaired and checked by reading, not compiled, not run on hardware
//
#include <hip/hip_runtime.h>
#include <hip/hip_bf16.h>
#include <math.h>

// MDAUnit fused block for gfx1250 (MI455X). All heavy GEMMs use
// v_wmma_f32_16x16x32_bf16 (wave32, fp32 accumulate). See analysis above.

typedef __attribute__((ext_vector_type(16))) __bf16 v16bf;
typedef __attribute__((ext_vector_type(8)))  float  v8f;

union ABf { v16bf v; uint4 q[2]; unsigned short u[16]; };
union Acc { v8f v; float f[8]; };

#define NE   8388608   // 32*64*64*64
#define BSTR 262144    // 64*64*64 per-batch stride

__device__ __forceinline__ unsigned short f2bf(float x) {
  union { float f; unsigned int u; } c; c.f = x;
  unsigned int u = c.u;
  u += 0x7fffu + ((u >> 16) & 1u);       // round-to-nearest-even
  return (unsigned short)(u >> 16);
}

__device__ __forceinline__ v8f wmma_bf(const ABf& a, const ABf& b, v8f c) {
  return __builtin_amdgcn_wmma_f32_16x16x32_bf16(
      /*neg_a=*/false, a.v, /*neg_b=*/false, b.v,
      /*c_mod=*/(short)0, c, /*reuse_a=*/false, /*reuse_b=*/false);
}

__device__ __forceinline__ float gelu_erf(float x) {
  return 0.5f * x * (1.0f + erff(x * 0.70710678118654752f));
}

// ---------- weight prep: fp32 -> bf16, optional transpose ----------
__global__ void k_convert_bf(const float* __restrict__ src,
                             unsigned short* __restrict__ dst, int n) {
  int i = blockIdx.x * blockDim.x + threadIdx.x;
  if (i < n) dst[i] = f2bf(src[i]);
}

// dst[c*R + r] = bf16(src[r*C + c])   (dst is the N-major / transposed copy)
__global__ void k_transpose_bf(const float* __restrict__ src,
                               unsigned short* __restrict__ dst, int R, int C) {
  int i = blockIdx.x * blockDim.x + threadIdx.x;
  if (i >= R * C) return;
  int c = i / R, r = i - c * R;
  dst[i] = f2bf(src[r * C + c]);
}

// ---------- conv1 1x1 + BN + ReLU, output in permuted (B,W,H,C) fp32 ----------
// Per block: fixed (b,h). M = w (64), N = c (64), K = 256.
__global__ void k_conv1(const float* __restrict__ x,
                        const unsigned short* __restrict__ c1w, // [c][k] bf16
                        const float* __restrict__ g, const float* __restrict__ bb,
                        const float* __restrict__ mm, const float* __restrict__ vv,
                        float* __restrict__ xp) {
  int b = blockIdx.x >> 6, h = blockIdx.x & 63;
  int t = threadIdx.x, wv = t >> 5, lane = t & 31;
  int nloc = lane & 15, half = lane >> 4;
  __shared__ __align__(16) unsigned short Alds[64 * 40];  // 64 rows, stride 40 (80B)

  int cidx = wv * 16 + nloc;
  float inv  = g[cidx] * rsqrtf(vv[cidx] + 1e-5f);
  float beta = bb[cidx] - mm[cidx] * inv;

  Acc acc[4];
  for (int mt = 0; mt < 4; mt++)
    for (int i = 0; i < 8; i++) acc[mt].f[i] = 0.f;

  for (int k0 = 0; k0 < 256; k0 += 32) {
    // stage A tile: Alds[w][kk] = bf16(x[b][k0+kk][h][w]); coalesced over w
    for (int e = 0; e < 16; e++) {
      int idx = e * 128 + t;
      int w_ = idx & 63, kk = idx >> 6;
      Alds[w_ * 40 + kk] = f2bf(x[b * 1048576 + (k0 + kk) * 4096 + h * 64 + w_]);
    }
    __syncthreads();
    // B fragment (fixed per wave): Bt[c][k] = c1w contiguous
    ABf bf;
    const uint4* bp = (const uint4*)(c1w + cidx * 256 + k0 + half * 16);
    bf.q[0] = bp[0]; bf.q[1] = bp[1];
    for (int mt = 0; mt < 4; mt++) {
      int row = mt * 16 + nloc;
      const unsigned short* ap = Alds + row * 40;
      ABf af;
      af.q[0] = *(const uint4*)(ap + half * 16);
      af.q[1] = *(const uint4*)(ap + half * 16 + 8);
      acc[mt].v = wmma_bf(af, bf, acc[mt].v);
    }
    __syncthreads();
  }
  for (int mt = 0; mt < 4; mt++)
    for (int r = 0; r < 8; r++) {
      int w_ = mt * 16 + r + half * 8;
      float val = fmaxf(acc[mt].f[r] * inv + beta, 0.f);
      xp[((b * 64 + w_) * 64 + h) * 64 + cidx] = val;
    }
}

// ---------- per-branch LayerNorm over C=64, output bf16 ----------
__global__ void k_ln(const float* __restrict__ xp,
                     const float* __restrict__ lng, const float* __restrict__ lnb,
                     unsigned short* __restrict__ xln) {
  int row = blockIdx.x * blockDim.x + threadIdx.x;   // 131072 rows
  const float* src = xp + (size_t)row * 64;
  float s = 0.f, s2 = 0.f;
  for (int c = 0; c < 64; c += 4) {
    float4 q = *(const float4*)(src + c);
    s  += q.x + q.y + q.z + q.w;
    s2 += q.x * q.x + q.y * q.y + q.z * q.z + q.w * q.w;
  }
  float mu = s * (1.f / 64.f);
  float var = s2 * (1.f / 64.f) - mu * mu;
  float rstd = rsqrtf(var + 1e-5f);
  unsigned int* dst = (unsigned int*)xln + (size_t)row * 32;
  for (int c = 0; c < 64; c += 2) {
    float y0 = (src[c]     - mu) * rstd * lng[c]     + lnb[c];
    float y1 = (src[c + 1] - mu) * rstd * lng[c + 1] + lnb[c + 1];
    dst[c >> 1] = (unsigned int)f2bf(y0) | ((unsigned int)f2bf(y1) << 16);
  }
}

// ---------- h-branch batched 512x512x512 GEMM (A direct from global) ----------
// rows r = g*64+j, cols q' = i*8+s;  A[r][K] = xln[b][K>>3][j][g*8+(K&7)]
__global__ void k_gemm_h(const unsigned short* __restrict__ xln,
                         const unsigned short* __restrict__ whT,  // [n][k] bf16
                         float* __restrict__ hbuf) {
  int b = blockIdx.x, rblk = blockIdx.y, nblk = blockIdx.z;
  int t = threadIdx.x, wv = t >> 5, lane = t & 31;
  int nloc = lane & 15, half = lane >> 4;
  const unsigned short* xb = xln + (size_t)b * BSTR;
  int R0 = rblk * 64 + wv * 16;
  int r = R0 + nloc;
  int gA = r >> 6, jA = r & 63;
  int N0 = nblk * 64;
  Acc acc[4];
  for (int nt = 0; nt < 4; nt++) for (int i = 0; i < 8; i++) acc[nt].f[i] = 0.f;

  for (int k0 = 0; k0 < 512; k0 += 32) {
    int i0 = (k0 + half * 16) >> 3;
    int i1 = (k0 + half * 16 + 8) >> 3;
    ABf af;
    af.q[0] = *(const uint4*)(xb + i0 * 4096 + jA * 64 + gA * 8);
    af.q[1] = *(const uint4*)(xb + i1 * 4096 + jA * 64 + gA * 8);
    for (int nt = 0; nt < 4; nt++) {
      int n = N0 + nt * 16 + nloc;
      const uint4* bp = (const uint4*)(whT + n * 512 + k0 + half * 16);
      ABf bf; bf.q[0] = bp[0]; bf.q[1] = bp[1];
      acc[nt].v = wmma_bf(af, bf, acc[nt].v);
    }
  }
  float* hb = hbuf + (size_t)b * BSTR;
  for (int nt = 0; nt < 4; nt++) {
    int q = N0 + nt * 16 + nloc;
    int iO = q >> 3, sO = q & 7;
    for (int rr = 0; rr < 8; rr++) {
      int row = R0 + rr + half * 8;
      int gO = row >> 6, jO = row & 63;
      hb[iO * 4096 + jO * 64 + gO * 8 + sO] = acc[nt].f[rr];
    }
  }
}

// ---------- w-branch batched 512x512x512 GEMM ----------
// rows r = i*8+g, cols q' = j*8+s;  A[r][K] = xln[b][i][K>>3][g*8+(K&7)]
__global__ void k_gemm_w(const unsigned short* __restrict__ xln,
                         const unsigned short* __restrict__ wwT,
                         float* __restrict__ wbuf) {
  int b = blockIdx.x, rblk = blockIdx.y, nblk = blockIdx.z;
  int t = threadIdx.x, wv = t >> 5, lane = t & 31;
  int nloc = lane & 15, half = lane >> 4;
  const unsigned short* xb = xln + (size_t)b * BSTR;
  int R0 = rblk * 64 + wv * 16;
  int r = R0 + nloc;
  int iA = r >> 3, gA = r & 7;
  int N0 = nblk * 64;
  Acc acc[4];
  for (int nt = 0; nt < 4; nt++) for (int i = 0; i < 8; i++) acc[nt].f[i] = 0.f;

  for (int k0 = 0; k0 < 512; k0 += 32) {
    int j0 = (k0 + half * 16) >> 3;
    int j1 = (k0 + half * 16 + 8) >> 3;
    ABf af;
    af.q[0] = *(const uint4*)(xb + iA * 4096 + j0 * 64 + gA * 8);
    af.q[1] = *(const uint4*)(xb + iA * 4096 + j1 * 64 + gA * 8);
    for (int nt = 0; nt < 4; nt++) {
      int n = N0 + nt * 16 + nloc;
      const uint4* bp = (const uint4*)(wwT + n * 512 + k0 + half * 16);
      ABf bf; bf.q[0] = bp[0]; bf.q[1] = bp[1];
      acc[nt].v = wmma_bf(af, bf, acc[nt].v);
    }
  }
  float* wb = wbuf + (size_t)b * BSTR;
  for (int nt = 0; nt < 4; nt++) {
    int q = N0 + nt * 16 + nloc;
    int jO = q >> 3, sO = q & 7;
    for (int rr = 0; rr < 8; rr++) {
      int row = R0 + rr + half * 8;
      int iO = row >> 3, gO = row & 7;
      wb[iO * 4096 + jO * 64 + gO * 8 + sO] = acc[nt].f[rr];
    }
  }
}

// ---------- c = xln @ wc : per block fixed (b,i); M=j(64), N=c(64), K=64 ----------
__global__ void k_gemm_c(const unsigned short* __restrict__ xln,
                         const unsigned short* __restrict__ wcT,  // [n][k]
                         float* __restrict__ cbuf) {
  int b = blockIdx.x >> 6, ii = blockIdx.x & 63;
  int t = threadIdx.x, wv = t >> 5, lane = t & 31;
  int nloc = lane & 15, half = lane >> 4;
  const unsigned short* xb = xln + (size_t)b * BSTR + ii * 4096;
  Acc acc[4];
  for (int mt = 0; mt < 4; mt++) for (int i = 0; i < 8; i++) acc[mt].f[i] = 0.f;

  for (int k0 = 0; k0 < 64; k0 += 32) {
    ABf bf;
    const uint4* bp = (const uint4*)(wcT + (wv * 16 + nloc) * 64 + k0 + half * 16);
    bf.q[0] = bp[0]; bf.q[1] = bp[1];
    for (int mt = 0; mt < 4; mt++) {
      int j = mt * 16 + nloc;
      ABf af;
      af.q[0] = *(const uint4*)(xb + j * 64 + k0 + half * 16);
      af.q[1] = *(const uint4*)(xb + j * 64 + k0 + half * 16 + 8);
      acc[mt].v = wmma_bf(af, bf, acc[mt].v);
    }
  }
  float* cb = cbuf + (size_t)b * BSTR + ii * 4096;
  int cc = wv * 16 + nloc;
  for (int mt = 0; mt < 4; mt++)
    for (int rr = 0; rr < 8; rr++) {
      int j = mt * 16 + rr + half * 8;
      cb[j * 64 + cc] = acc[mt].f[rr];
    }
}

// ---------- means + tiny GELU MLPs ----------
// blockIdx.y==0: avec[b][j][c] = mlp_h(mean_i h);  ==1: bvec[b][i][c] = mlp_w(mean_j w)
__global__ void k_reduce_mlp(const float* __restrict__ hbuf, const float* __restrict__ wbuf,
                             const float* __restrict__ mh1, const float* __restrict__ mh2,
                             const float* __restrict__ mw1, const float* __restrict__ mw2,
                             float* __restrict__ avec, float* __restrict__ bvec) {
  int which = blockIdx.y;
  int b = blockIdx.x >> 6, rj = blockIdx.x & 63;
  int c = threadIdx.x;
  float s = 0.f;
  if (which == 0) {
    size_t base = (size_t)b * BSTR + rj * 64 + c;           // mean over i (stride 4096)
    for (int i = 0; i < 64; i++) s += hbuf[base + i * 4096];
  } else {
    size_t base = (size_t)b * BSTR + rj * 4096 + c;         // mean over j (stride 64)
    for (int j = 0; j < 64; j++) s += wbuf[base + j * 64];
  }
  s *= (1.f / 64.f);
  __shared__ float v0[64];
  __shared__ float h1[16];
  v0[c] = s;
  __syncthreads();
  const float* m1 = which ? mw1 : mh1;
  const float* m2 = which ? mw2 : mh2;
  if (c < 16) {
    float tv = 0.f;
    for (int k = 0; k < 64; k++) tv += v0[k] * m1[k * 16 + c];
    h1[c] = gelu_erf(tv);
  }
  __syncthreads();
  float tv = 0.f;
  for (int k = 0; k < 16; k++) tv += h1[k] * m2[k * 64 + c];
  float o = gelu_erf(tv);
  if (which == 0) avec[b * 4096 + rj * 64 + c] = o;
  else            bvec[b * 4096 + rj * 64 + c] = o;
}

// ---------- combine: out = h*bv + w*av + c*(av+bv), write bf16 ----------
__global__ void k_combine(const float* __restrict__ hbuf, const float* __restrict__ wbuf,
                          const float* __restrict__ cbuf,
                          const float* __restrict__ avec, const float* __restrict__ bvec,
                          unsigned short* __restrict__ outb) {
  int idx = blockIdx.x * blockDim.x + threadIdx.x;
  if (idx >= NE) return;
  int c = idx & 63, j = (idx >> 6) & 63, i = (idx >> 12) & 63, b = idx >> 18;
  float av = avec[b * 4096 + j * 64 + c];
  float bv = bvec[b * 4096 + i * 64 + c];
  float o = hbuf[idx] * bv + wbuf[idx] * av + cbuf[idx] * (av + bv);
  outb[idx] = f2bf(o);
}

// ---------- conv3 1x1 + BN + residual + ReLU ----------
// Per block: fixed (b,h,oblk). M = o (64 rows of 256), N = w (64), K = 128.
__global__ void k_conv3(const unsigned short* __restrict__ c3w,    // [o][k] bf16
                        const unsigned short* __restrict__ out1b,  // k<64  (branch b_*)
                        const unsigned short* __restrict__ out2b,  // k>=64 (branch a_*)
                        const float* __restrict__ xin,
                        const float* __restrict__ g, const float* __restrict__ bb,
                        const float* __restrict__ mm, const float* __restrict__ vv,
                        float* __restrict__ dout) {
  int b = blockIdx.x >> 6, h = blockIdx.x & 63;
  int oblk = blockIdx.y;
  int t = threadIdx.x, wv = t >> 5, lane = t & 31;
  int nloc = lane & 15, half = lane >> 4;
  int O0 = oblk * 64 + wv * 16;
  int orow = O0 + nloc;
  Acc acc[4];
  for (int nt = 0; nt < 4; nt++) for (int i = 0; i < 8; i++) acc[nt].f[i] = 0.f;

  for (int k0 = 0; k0 < 128; k0 += 32) {
    ABf af;
    af.q[0] = *(const uint4*)(c3w + orow * 128 + k0 + half * 16);
    af.q[1] = *(const uint4*)(c3w + orow * 128 + k0 + half * 16 + 8);
    int Kb = k0 + half * 16;
    const unsigned short* src = (Kb < 64) ? out1b : out2b;
    int koff = Kb & 63;
    for (int nt = 0; nt < 4; nt++) {
      int wcol = nt * 16 + nloc;
      const uint4* bp = (const uint4*)(src + ((size_t)(b * 64 + wcol) * 64 + h) * 64 + koff);
      ABf bf; bf.q[0] = bp[0]; bf.q[1] = bp[1];
      acc[nt].v = wmma_bf(af, bf, acc[nt].v);
    }
  }
  for (int nt = 0; nt < 4; nt++) {
    int wcol = nt * 16 + nloc;
    for (int rr = 0; rr < 8; rr++) {
      int o = O0 + rr + half * 8;
      float inv  = g[o] * rsqrtf(vv[o] + 1e-5f);
      float beta = bb[o] - mm[o] * inv;
      size_t idx = ((size_t)(b * 256 + o) * 64 + h) * 64 + wcol;
      float val = acc[nt].f[rr] * inv + beta + xin[idx];
      dout[idx] = fmaxf(val, 0.f);
    }
  }
}

extern "C" void kernel_launch(void* const* d_in, const int* in_sizes, int n_in,
                              void* d_out, int out_size, void* d_ws, size_t ws_size,
                              hipStream_t stream) {
  (void)in_sizes; (void)n_in; (void)out_size; (void)ws_size;
  const float* X    = (const float*)d_in[0];
  const float* C1W  = (const float*)d_in[1];
  const float* BN1G = (const float*)d_in[2];
  const float* BN1B = (const float*)d_in[3];
  const float* BN1M = (const float*)d_in[4];
  const float* BN1V = (const float*)d_in[5];
  const float* C3W  = (const float*)d_in[6];
  const float* BN3G = (const float*)d_in[7];
  const float* BN3B = (const float*)d_in[8];
  const float* BN3M = (const float*)d_in[9];
  const float* BN3V = (const float*)d_in[10];
  const float* ALNG = (const float*)d_in[11];
  const float* ALNB = (const float*)d_in[12];
  const float* AWH  = (const float*)d_in[13];
  const float* AWW  = (const float*)d_in[14];
  const float* AWC  = (const float*)d_in[15];
  const float* AMH1 = (const float*)d_in[16];
  const float* AMH2 = (const float*)d_in[17];
  const float* AMW1 = (const float*)d_in[18];
  const float* AMW2 = (const float*)d_in[19];
  const float* BLNG = (const float*)d_in[20];
  const float* BLNB = (const float*)d_in[21];
  const float* BWH  = (const float*)d_in[22];
  const float* BWW  = (const float*)d_in[23];
  const float* BWC  = (const float*)d_in[24];
  const float* BMH1 = (const float*)d_in[25];
  const float* BMH2 = (const float*)d_in[26];
  const float* BMW1 = (const float*)d_in[27];
  const float* BMW2 = (const float*)d_in[28];

  // ---- workspace carve (256B-aligned slabs) ----
  char* p = (char*)d_ws;
  auto take = [&](size_t bytes) -> void* {
    void* r = (void*)p;
    p += (bytes + 255) & ~(size_t)255;
    return r;
  };
  float*          xp    = (float*)take((size_t)NE * 4);
  unsigned short* xln   = (unsigned short*)take((size_t)NE * 2);
  float*          hbuf  = (float*)take((size_t)NE * 4);
  float*          wbuf  = (float*)take((size_t)NE * 4);
  float*          cbuf  = (float*)take((size_t)NE * 4);
  float*          avec  = (float*)take(131072 * 4);
  float*          bvec  = (float*)take(131072 * 4);
  unsigned short* out1b = (unsigned short*)take((size_t)NE * 2);
  unsigned short* out2b = (unsigned short*)take((size_t)NE * 2);
  unsigned short* c1wb  = (unsigned short*)take(64 * 256 * 2);
  unsigned short* c3wb  = (unsigned short*)take(256 * 128 * 2);
  unsigned short* whTa  = (unsigned short*)take(512 * 512 * 2);
  unsigned short* wwTa  = (unsigned short*)take(512 * 512 * 2);
  unsigned short* whTb  = (unsigned short*)take(512 * 512 * 2);
  unsigned short* wwTb  = (unsigned short*)take(512 * 512 * 2);
  unsigned short* wcTa  = (unsigned short*)take(64 * 64 * 2);
  unsigned short* wcTb  = (unsigned short*)take(64 * 64 * 2);

  // ---- weight prep ----
  k_convert_bf<<<64, 256, 0, stream>>>(C1W, c1wb, 64 * 256);
  k_convert_bf<<<128, 256, 0, stream>>>(C3W, c3wb, 256 * 128);
  k_transpose_bf<<<1024, 256, 0, stream>>>(AWH, whTa, 512, 512);
  k_transpose_bf<<<1024, 256, 0, stream>>>(AWW, wwTa, 512, 512);
  k_transpose_bf<<<1024, 256, 0, stream>>>(BWH, whTb, 512, 512);
  k_transpose_bf<<<1024, 256, 0, stream>>>(BWW, wwTb, 512, 512);
  k_transpose_bf<<<16, 256, 0, stream>>>(AWC, wcTa, 64, 64);
  k_transpose_bf<<<16, 256, 0, stream>>>(BWC, wcTb, 64, 64);

  // ---- conv1 + BN + ReLU -> xp (B,W,H,C) fp32 ----
  k_conv1<<<2048, 128, 0, stream>>>(X, c1wb, BN1G, BN1B, BN1M, BN1V, xp);

  // ---- branch A (a_* params) -> out2b ----
  k_ln<<<1024, 128, 0, stream>>>(xp, ALNG, ALNB, xln);
  k_gemm_h<<<dim3(32, 8, 8), 128, 0, stream>>>(xln, whTa, hbuf);
  k_gemm_w<<<dim3(32, 8, 8), 128, 0, stream>>>(xln, wwTa, wbuf);
  k_gemm_c<<<2048, 128, 0, stream>>>(xln, wcTa, cbuf);
  k_reduce_mlp<<<dim3(2048, 2), 64, 0, stream>>>(hbuf, wbuf, AMH1, AMH2, AMW1, AMW2,
                                                 avec, bvec);
  k_combine<<<NE / 256, 256, 0, stream>>>(hbuf, wbuf, cbuf, avec, bvec, out2b);

  // ---- branch B (b_* params) -> out1b ----
  k_ln<<<1024, 128, 0, stream>>>(xp, BLNG, BLNB, xln);
  k_gemm_h<<<dim3(32, 8, 8), 128, 0, stream>>>(xln, whTb, hbuf);
  k_gemm_w<<<dim3(32, 8, 8), 128, 0, stream>>>(xln, wwTb, wbuf);
  k_gemm_c<<<2048, 128, 0, stream>>>(xln, wcTb, cbuf);
  k_reduce_mlp<<<dim3(2048, 2), 64, 0, stream>>>(hbuf, wbuf, BMH1, BMH2, BMW1, BMW2,
                                                 avec, bvec);
  k_combine<<<NE / 256, 256, 0, stream>>>(hbuf, wbuf, cbuf, avec, bvec, out1b);

  // ---- conv3 + BN + residual + ReLU -> d_out (B,256,64,64) fp32 ----
  k_conv3<<<dim3(2048, 4), 128, 0, stream>>>(c3wb, out1b, out2b, X,
                                             BN3G, BN3B, BN3M, BN3V, (float*)d_out);
}
